// EpisodicSDM_61538291417812
// MI455X (gfx1250) — matrix operand order repaired and checked by reading, not compiled, not run on hardware
//
#include <hip/hip_runtime.h>
#include <math.h>

typedef __attribute__((ext_vector_type(16))) _Float16 v16h;
typedef __attribute__((ext_vector_type(8)))  _Float16 v8h;
typedef __attribute__((ext_vector_type(8)))  float    v8f;
typedef __attribute__((ext_vector_type(4)))  unsigned int u32x4;
typedef __attribute__((ext_vector_type(8)))  int          i32x8;
typedef __attribute__((ext_vector_type(4)))  int          i32x4;

#define DIM 256
#define B_ROWS 2048
#define N_KEYS 100000
#define TOPK 8
#define NSLICES 25
#define CHUNKS_PER_SLICE 250           /* 25 * 250 = 6250 chunks of 16 keys */
#define CAND_PER_ROW (NSLICES * 16)    /* 400 candidates per row */
#define CAND_PAD 416                   /* 32 lanes * 13 slots */
#define ROWS_PER_BLOCK 128             /* 4 waves * 32 rows */
#define LDS_ROW_STRIDE 528             /* 512B key row + 16B pad (bank spread) */
#define LDS_TILE (16 * LDS_ROW_STRIDE) /* 8448 B per 16-key tile */
#define NBUF 4                         /* ring of staged key tiles */

#if defined(__HIP_DEVICE_COMPILE__) && __has_builtin(__builtin_amdgcn_tensor_load_to_lds) && __has_builtin(__builtin_amdgcn_s_wait_tensorcnt)
#define USE_TDM 1
#else
#define USE_TDM 0
#endif

// ---------------------------------------------------------------------------
// Kernel 1: row-wise L2 normalize f32 -> f16 (one 256-thread block per row)
// ---------------------------------------------------------------------------
__global__ __launch_bounds__(256) void rownorm_f16_kernel(
    const float* __restrict__ src, _Float16* __restrict__ dst) {
  __shared__ float red[256];
  const int r = blockIdx.x;
  const int d = threadIdx.x;
  const float v = src[(size_t)r * DIM + d];
  red[d] = v * v;
  __syncthreads();
  #pragma unroll
  for (int s = 128; s > 0; s >>= 1) {
    if (d < s) red[d] += red[d + s];
    __syncthreads();
  }
  const float inv = 1.0f / fmaxf(sqrtf(red[0]), 1e-12f);
  dst[(size_t)r * DIM + d] = (_Float16)(v * inv);
}

// ---------------------------------------------------------------------------
// TDM staging: one descriptor moves a 16-key x 512B tile into LDS, inserting
// 4 dwords of pad every 128 dwords (row stride 528B). Issued by wave 0 only;
// tracked with TENSORcnt (in-order completion per wave).
// ---------------------------------------------------------------------------
#if USE_TDM
__device__ __forceinline__ void tdm_stage(const _Float16* gsrc, unsigned ldsAddr) {
  const unsigned long long ga = (unsigned long long)gsrc;
  u32x4 g0;
  g0.x = 1u;                                   // count=1 (valid user descriptor)
  g0.y = ldsAddr;                              // lds_addr (bytes)
  g0.z = (unsigned)(ga & 0xFFFFFFFFu);         // global_addr[31:0]
  g0.w = (unsigned)((ga >> 32) & 0x01FFFFFFu)  // global_addr[56:32]
         | (2u << 30);                         // type = 2 ("image")
  i32x8 g1;
  g1[0] = (2 << 16)        // data_size = 4B
        | (1 << 20)        // pad_enable
        | (6 << 22)        // pad_interval: 2^(6+1)=128 dwords (one 512B row)
        | (3 << 25);       // pad_amount: 4 dwords (16B)
  g1[1] = (int)(128u << 16);   // tensor_dim0 = 128 dwords (bits[79:48] low half)
  g1[2] = (int)(16u  << 16);   // tensor_dim1 = 16 rows   (bits[111:80] low half)
  g1[3] = (int)(128u << 16);   // tile_dim0 = 128 dwords  (bits[127:112])
  g1[4] = 16;                  // tile_dim1 = 16 rows     (bits[143:128])
  g1[5] = 128;                 // tensor_dim0_stride = 128 dwords
  g1[6] = 0;
  g1[7] = 0;
  const i32x4 z4 = {0, 0, 0, 0};
  const i32x8 z8 = {0, 0, 0, 0, 0, 0, 0, 0};
  __builtin_amdgcn_tensor_load_to_lds(g0, g1, z4, z4, z8, 0);
}
#else
// Fallback: cooperative global->LDS copy by all 128 threads (4 x 16B each).
__device__ __forceinline__ void coop_stage(const _Float16* __restrict__ gsrc,
                                           unsigned char* __restrict__ dst, int tid) {
  #pragma unroll
  for (int j = 0; j < 4; ++j) {
    const int p   = tid + 128 * j;       // 512 pieces of 16B
    const int row = p >> 5;
    const int col = p & 31;
    *(v8h*)(dst + row * LDS_ROW_STRIDE + col * 16) =
        *(const v8h*)(gsrc + row * DIM + col * 8);
  }
}
#endif

// ---------------------------------------------------------------------------
// Per-lane running top-8 update from one 8-candidate accumulator column.
// ---------------------------------------------------------------------------
__device__ __forceinline__ void topk_update(const v8f& c, int keyBase,
                                            float (&tv)[TOPK], int (&ti)[TOPK],
                                            float& vmin) {
  float cmax = c[0];
  #pragma unroll
  for (int j = 1; j < 8; ++j) cmax = fmaxf(cmax, c[j]);
  if (cmax > vmin) {                    // rare after warm-up
    #pragma unroll
    for (int j = 0; j < 8; ++j) {
      const float v = c[j];
      if (v > vmin) {
        bool done = false;
        #pragma unroll
        for (int s = 0; s < TOPK; ++s) {
          const bool repl = (!done) && (tv[s] == vmin);
          tv[s] = repl ? v : tv[s];
          ti[s] = repl ? (keyBase + j) : ti[s];
          done = done || repl;
        }
        float m = tv[0];
        #pragma unroll
        for (int s = 1; s < TOPK; ++s) m = fminf(m, tv[s]);
        vmin = m;
      }
    }
  }
}

// ---------------------------------------------------------------------------
// Kernel 2: fused scores GEMM + top-8. Block = 4 waves x 32 query rows.
// Key chunks staged by the TDM into a 4-deep LDS ring, issued 3 chunks
// ahead (s_wait_tensorcnt <= 2 before consuming, since TDM completes
// in order). All 4 waves consume each staged tile -> 4x key-traffic
// reduction vs private streaming.
// ---------------------------------------------------------------------------
__global__ __launch_bounds__(128) void sdm_scores_topk_kernel(
    const _Float16* __restrict__ xf, const _Float16* __restrict__ kf,
    float* __restrict__ pvals, int* __restrict__ pidx) {
  __shared__ __align__(16) unsigned char smem[NBUF * LDS_TILE];

  const int tid     = threadIdx.x;
  const int lane    = tid & 31;
  const int laneRow = lane & 15;
  const int hi16    = lane >> 4;
  const int wave    = tid >> 5;
  const int rowBase = blockIdx.x * ROWS_PER_BLOCK + wave * 32;
  const int slice   = blockIdx.y;

  // Persistent B operands: 32 x-rows (two 16-row tiles), 128 VGPRs.
  v16h xb0[8], xb1[8];
  {
    const _Float16* xp0 = xf + (size_t)(rowBase + laneRow) * DIM + hi16 * 16;
    const _Float16* xp1 = xp0 + 16 * DIM;
    #pragma unroll
    for (int kb = 0; kb < 8; ++kb) {
      xb0[kb] = *(const v16h*)(xp0 + kb * 32);
      xb1[kb] = *(const v16h*)(xp1 + kb * 32);
    }
  }

  float tv0[TOPK], tv1[TOPK]; int ti0[TOPK], ti1[TOPK];
  #pragma unroll
  for (int s = 0; s < TOPK; ++s) {
    tv0[s] = -3.0e38f; ti0[s] = 0;
    tv1[s] = -3.0e38f; ti1[s] = 0;
  }
  float vmin0 = -3.0e38f, vmin1 = -3.0e38f;

  const int c0s = slice * CHUNKS_PER_SLICE;
  const int c1s = c0s + CHUNKS_PER_SLICE;

  // Prologue: stage chunks c0s .. c0s+2 into ring buffers 0..2.
#if USE_TDM
  const unsigned ldsBase = (unsigned)(unsigned long long)&smem[0];
  if (tid < 32) {
    #pragma unroll
    for (int j = 0; j < 3; ++j)
      tdm_stage(kf + (size_t)(c0s + j) * (16 * DIM), ldsBase + j * LDS_TILE);
  }
#else
  #pragma unroll
  for (int j = 0; j < 3; ++j)
    coop_stage(kf + (size_t)(c0s + j) * (16 * DIM), smem + j * LDS_TILE, tid);
#endif

  for (int chunk = c0s; chunk < c1s; ++chunk) {
    const int bi = (chunk - c0s) & (NBUF - 1);

#if USE_TDM
    if (tid < 32) {                     // ensure buffer `bi` has landed
      const int rem = c1s - 1 - chunk;  // uniform; immediates required
      if (rem >= 2)      __builtin_amdgcn_s_wait_tensorcnt(2);
      else if (rem == 1) __builtin_amdgcn_s_wait_tensorcnt(1);
      else               __builtin_amdgcn_s_wait_tensorcnt(0);
    }
#endif
    __syncthreads();                    // publish `bi`; ring slot (bi+3)&3 drained

#if USE_TDM
    if ((tid < 32) && (chunk + 3 < c1s))
      tdm_stage(kf + (size_t)(chunk + 3) * (16 * DIM),
                ldsBase + ((bi + 3) & (NBUF - 1)) * LDS_TILE);
#else
    if (chunk + 3 < c1s)
      coop_stage(kf + (size_t)(chunk + 3) * (16 * DIM),
                 smem + ((bi + 3) & (NBUF - 1)) * LDS_TILE, tid);
#endif

    // Materialize all 8 A fragments first (distinct regs -> overlapped ds).
    const unsigned char* abase =
        smem + bi * LDS_TILE + laneRow * LDS_ROW_STRIDE + hi16 * 32;
    v16h a[8];
    #pragma unroll
    for (int kb = 0; kb < 8; ++kb) {
      v8h lo = *(const v8h*)(abase + kb * 64);
      v8h hi = *(const v8h*)(abase + kb * 64 + 16);
      a[kb] = __builtin_shufflevector(lo, hi,
                0,1,2,3,4,5,6,7,8,9,10,11,12,13,14,15);
    }

    v8f cA = {}, cB = {};
    #pragma unroll
    for (int kb = 0; kb < 8; ++kb) {
      cA = __builtin_amdgcn_wmma_f32_16x16x32_f16(
              false, a[kb], false, xb0[kb], (short)0, cA, false, false);
      cB = __builtin_amdgcn_wmma_f32_16x16x32_f16(
              false, a[kb], false, xb1[kb], (short)0, cB, false, false);
    }

    const int keyBase = chunk * 16 + hi16 * 8;
    topk_update(cA, keyBase, tv0, ti0, vmin0);
    topk_update(cB, keyBase, tv1, ti1, vmin1);
  }

  // Emit per-lane top-8 for both owned rows (order-free; merged later).
  const size_t b0 = ((size_t)(rowBase + laneRow) * NSLICES + slice) * 16 + hi16 * 8;
  const size_t b1 = ((size_t)(rowBase + 16 + laneRow) * NSLICES + slice) * 16 + hi16 * 8;
  #pragma unroll
  for (int s = 0; s < TOPK; ++s) {
    pvals[b0 + s] = tv0[s]; pidx[b0 + s] = ti0[s];
    pvals[b1 + s] = tv1[s]; pidx[b1 + s] = ti1[s];
  }
}

// ---------------------------------------------------------------------------
// Kernel 3: per row merge 400 candidates -> top-8, softmax, gather + blend.
// ---------------------------------------------------------------------------
__global__ __launch_bounds__(256) void sdm_finalize_kernel(
    const float* __restrict__ pvals, const int* __restrict__ pidx,
    const float* __restrict__ values, float* __restrict__ out) {
  __shared__ float cv[CAND_PAD];
  __shared__ int   ci[CAND_PAD];
  __shared__ float selV[TOPK];
  __shared__ int   selI[TOPK];
  __shared__ float w[TOPK];

  const int row = blockIdx.x;
  const int t = threadIdx.x;
  for (int i = t; i < CAND_PAD; i += 256) {
    const bool inr = i < CAND_PER_ROW;
    cv[i] = inr ? pvals[(size_t)row * CAND_PER_ROW + i] : -3.0e38f;
    ci[i] = inr ? pidx[(size_t)row * CAND_PER_ROW + i] : 0;
  }
  __syncthreads();

  if (t < 32) {                 // wave 0: 8 rounds of wave-wide argmax over 416
    float v[13]; int id[13];
    #pragma unroll
    for (int s = 0; s < 13; ++s) { const int p = t * 13 + s; v[s] = cv[p]; id[s] = ci[p]; }
    for (int k = 0; k < TOPK; ++k) {
      float lm = v[0]; int ls = 0;
      #pragma unroll
      for (int s = 1; s < 13; ++s) if (v[s] > lm) { lm = v[s]; ls = s; }
      float bm = lm; int bl = t;
      #pragma unroll
      for (int off = 16; off > 0; off >>= 1) {
        const float ov = __shfl_xor(bm, off, 32);
        const int   ol = __shfl_xor(bl, off, 32);
        if (ov > bm || (ov == bm && ol < bl)) { bm = ov; bl = ol; }
      }
      if (t == bl) {            // winning lane records and retires its slot
        int sidx = id[0];
        #pragma unroll
        for (int s = 1; s < 13; ++s) sidx = (s == ls) ? id[s] : sidx;
        selV[k] = lm; selI[k] = sidx;
        #pragma unroll
        for (int s = 0; s < 13; ++s) if (s == ls) v[s] = -3.0e38f;
      }
    }
  }
  __syncthreads();

  if (t == 0) {                 // softmax over the 8 selected scores
    float m = selV[0];
    #pragma unroll
    for (int k = 1; k < TOPK; ++k) m = fmaxf(m, selV[k]);
    float e[TOPK]; float sum = 0.f;
    #pragma unroll
    for (int k = 0; k < TOPK; ++k) { e[k] = __expf(selV[k] - m); sum += e[k]; }
    const float inv = 1.0f / sum;
    #pragma unroll
    for (int k = 0; k < TOPK; ++k) w[k] = e[k] * inv;
  }
  __syncthreads();

  float acc = 0.f;
  #pragma unroll
  for (int k = 0; k < TOPK; ++k)
    acc += w[k] * values[(size_t)selI[k] * DIM + t];
  out[(size_t)row * DIM + t] = acc;
}

// ---------------------------------------------------------------------------
extern "C" void kernel_launch(void* const* d_in, const int* in_sizes, int n_in,
                              void* d_out, int out_size, void* d_ws, size_t ws_size,
                              hipStream_t stream) {
  (void)in_sizes; (void)n_in; (void)out_size; (void)ws_size;
  const float* x      = (const float*)d_in[0];
  const float* keys   = (const float*)d_in[1];
  const float* values = (const float*)d_in[2];
  // d_in[3] = top_k scalar (fixed 8)
  float* out = (float*)d_out;

  char* ws = (char*)d_ws;
  const size_t xf_bytes = (size_t)B_ROWS * DIM * sizeof(_Float16);   // 1 MiB
  const size_t kf_bytes = (size_t)N_KEYS * DIM * sizeof(_Float16);   // 51.2 MB (L2-resident)
  const size_t pv_bytes = (size_t)B_ROWS * CAND_PER_ROW * sizeof(float);
  _Float16* xf    = (_Float16*)ws;
  _Float16* kf    = (_Float16*)(ws + xf_bytes);
  float*    pvals = (float*)   (ws + xf_bytes + kf_bytes);
  int*      pidx  = (int*)     (ws + xf_bytes + kf_bytes + pv_bytes);

  rownorm_f16_kernel<<<B_ROWS, 256, 0, stream>>>(x, xf);
  rownorm_f16_kernel<<<N_KEYS, 256, 0, stream>>>(keys, kf);

  dim3 grid(B_ROWS / ROWS_PER_BLOCK, NSLICES);   // 16 x 25
  sdm_scores_topk_kernel<<<grid, 128, 0, stream>>>(xf, kf, pvals, pidx);

  sdm_finalize_kernel<<<B_ROWS, 256, 0, stream>>>(pvals, pidx, values, out);
}
